// KVMoEModel_42116449304968
// MI455X (gfx1250) — matrix operand order repaired and checked by reading, not compiled, not run on hardware
//
#include <hip/hip_runtime.h>
#include <hip/hip_bf16.h>
#include <math.h>

#define BT 4096   // B*T tokens
#define C  128
#define E  16
#define KB 32
#define V  50257

#define TT 32     // tokens per attention block
#define PP 132    // padded LDS pitch (floats): 128 data + 4 pad (TDM pad_amount)

typedef __bf16 bf16_t;
typedef bf16_t v8bf  __attribute__((ext_vector_type(8)));
typedef bf16_t v16bf __attribute__((ext_vector_type(16)));
typedef float  v8f   __attribute__((ext_vector_type(8)));

typedef unsigned int u32x4 __attribute__((ext_vector_type(4)));
typedef int          i32x8 __attribute__((ext_vector_type(8)));
typedef int          i32x4 __attribute__((ext_vector_type(4)));

// ---------------- workspace layout (bytes) ----------------
constexpr size_t OFS_X    = 0;
constexpr size_t OFS_XLN  = OFS_X    + (size_t)BT * C * 4;
constexpr size_t OFS_RW   = OFS_XLN  + (size_t)BT * C * 4;
constexpr size_t OFS_Q    = OFS_RW   + (size_t)BT * E * 4;
constexpr size_t OFS_ATT  = OFS_Q    + (size_t)BT * C * 4;
constexpr size_t OFS_XRES = OFS_ATT  + (size_t)BT * C * 4;
constexpr size_t OFS_X2B  = OFS_XRES + (size_t)BT * C * 4;
constexpr size_t OFS_WLMB = OFS_X2B  + (size_t)BT * C * 2;

__device__ __forceinline__ unsigned short f2bf_rne(float f) {
  unsigned int u = __float_as_uint(f);
  unsigned int r = u + 0x7FFFu + ((u >> 16) & 1u);   // round-to-nearest-even
  return (unsigned short)(r >> 16);
}

// ---- TDM: DMA one 32x128 f32 tile (row stride 128) into LDS with 4-DW padding
// ---- per 128-DW row => LDS pitch of 132 floats. D# packing per ISA 8.3/8.4.
__device__ __forceinline__ void tdm_load_tile(const float* gsrc, unsigned lds_off) {
  const unsigned long long ga = (unsigned long long)(uintptr_t)gsrc;
  u32x4 g0;
  g0[0] = 1u;                                            // count=1, user desc
  g0[1] = lds_off;                                       // lds_addr [63:32]
  g0[2] = (unsigned)(ga & 0xFFFFFFFFu);                  // global_addr lo
  g0[3] = (unsigned)((ga >> 32) & 0x01FFFFFFu)           // global_addr [56:32]
          | 0x80000000u;                                 // type=2 ("image")
  i32x8 g1;
  g1[0] = (2 << 16)      // data_size = 4 bytes
        | (1 << 20)      // pad_enable
        | (6 << 22)      // pad_interval: every 2^(6+1)=128 DWORDs
        | (3 << 25);     // pad_amount: 4 DWORDs
  g1[1] = (int)(128u << 16);   // tensor_dim0 = 128 (bits [79:48] lo half)
  g1[2] = (int)(32u  << 16);   // tensor_dim1 = 32  (bits [111:80] lo half)
  g1[3] = (int)(128u << 16);   // tile_dim0 = 128   (bits [127:112])
  g1[4] = 32;                  // tile_dim1 = 32    (bits [143:128])
  g1[5] = 128;                 // tensor_dim0_stride = 128 (bits [207:160] lo)
  g1[6] = 0;
  g1[7] = 0;
  const i32x4 z4 = {0, 0, 0, 0};
  const i32x8 z8 = {0, 0, 0, 0, 0, 0, 0, 0};
  // 6-arg variant (clang-23 / therock-10.0 headers)
  __builtin_amdgcn_tensor_load_to_lds(g0, g1, z4, z4, z8, 0);
}

// ------------- kernel 1: embed gather + LN1 + router logits + softmax -------------
__global__ void k_embed_ln_gate(const int* __restrict__ ids,
                                const float* __restrict__ emb,
                                const float* __restrict__ router,
                                const float* __restrict__ g1,
                                const float* __restrict__ b1,
                                float* __restrict__ x,
                                float* __restrict__ xln,
                                float* __restrict__ rw,
                                float* __restrict__ gate_out) {
  const int t = blockIdx.x;
  const int c = threadIdx.x;
  __shared__ float red[C];
  __shared__ float logit[E];

  const int id = ids[t];
  const float v = emb[(size_t)id * C + c];
  x[(size_t)t * C + c] = v;

  red[c] = v; __syncthreads();
  for (int s = C / 2; s > 0; s >>= 1) { if (c < s) red[c] += red[c + s]; __syncthreads(); }
  const float mu = red[0] * (1.0f / C);
  __syncthreads();
  const float dv = v - mu;
  red[c] = dv * dv; __syncthreads();
  for (int s = C / 2; s > 0; s >>= 1) { if (c < s) red[c] += red[c + s]; __syncthreads(); }
  const float rstd = rsqrtf(red[0] * (1.0f / C) + 1e-5f);
  __syncthreads();
  const float xl = dv * rstd * g1[c] + b1[c];
  xln[(size_t)t * C + c] = xl;

  for (int e = 0; e < E; ++e) {
    red[c] = xl * router[e * C + c]; __syncthreads();
    for (int s = C / 2; s > 0; s >>= 1) { if (c < s) red[c] += red[c + s]; __syncthreads(); }
    if (c == 0) logit[e] = red[0];
    __syncthreads();
  }
  if (c == 0) {
    float m = logit[0];
    for (int e = 1; e < E; ++e) m = fmaxf(m, logit[e]);
    float s = 0.f, ex[E];
    for (int e = 0; e < E; ++e) { ex[e] = expf(logit[e] - m); s += ex[e]; }
    const float inv = 1.0f / s;
    for (int e = 0; e < E; ++e) {
      gate_out[(size_t)t * E + e] = logit[e];
      rw[(size_t)t * E + e] = ex[e] * inv;
    }
  }
}

// ------------- generic per-token 128x128 projection: out = in@W^T (+bias)(+resid) -------------
__global__ void k_rowmat(const float* __restrict__ in, const float* __restrict__ W,
                         const float* __restrict__ resid, const float* __restrict__ bias,
                         float* __restrict__ out) {
  const int t = blockIdx.x, d = threadIdx.x;
  __shared__ float row[C];
  row[d] = in[(size_t)t * C + d];
  __syncthreads();
  float acc = 0.f;
  const float* w = W + (size_t)d * C;
#pragma unroll 8
  for (int cc = 0; cc < C; ++cc) acc = fmaf(row[cc], w[cc], acc);
  if (bias)  acc += bias[d];
  if (resid) acc += resid[(size_t)t * C + d];
  out[(size_t)t * C + d] = acc;
}

// ------------- layernorm (for LN2) -------------
__global__ void k_layernorm(const float* __restrict__ in, const float* __restrict__ g,
                            const float* __restrict__ b, float* __restrict__ out) {
  const int t = blockIdx.x, c = threadIdx.x;
  __shared__ float red[C];
  const float v = in[(size_t)t * C + c];
  red[c] = v; __syncthreads();
  for (int s = C / 2; s > 0; s >>= 1) { if (c < s) red[c] += red[c + s]; __syncthreads(); }
  const float mu = red[0] * (1.0f / C);
  __syncthreads();
  const float dv = v - mu;
  red[c] = dv * dv; __syncthreads();
  for (int s = C / 2; s > 0; s >>= 1) { if (c < s) red[c] += red[c + s]; __syncthreads(); }
  const float rstd = rsqrtf(red[0] * (1.0f / C) + 1e-5f);
  out[(size_t)t * C + c] = dv * rstd * g[c] + b[c];
}

// ------------- expert-mixed KV attention, fused; expert tiles staged by TDM -------------
// Double-buffered: wave 0 issues tensor_load_to_lds for expert e+1, then
// s_wait_tensorcnt 1 (in-order TDM completion => tile e has landed), so the
// DMA for e+1 overlaps the FLOPs for e across the whole block.
__global__ __launch_bounds__(256) void k_attn(const float* __restrict__ q,
                                              const float* __restrict__ rw,
                                              const float* __restrict__ ke,
                                              const float* __restrict__ ve,
                                              const float* __restrict__ sk,
                                              const float* __restrict__ sv,
                                              const float* __restrict__ kvg,
                                              float* __restrict__ attn_out) {
  __shared__ __attribute__((aligned(16))) float qs[TT * PP];
  __shared__ __attribute__((aligned(16))) float es[2][KB * PP];
  __shared__ float rws[TT * E];
  __shared__ float sc[TT * (KB + 1)];

  const int tid = threadIdx.x;
  const int t0  = blockIdx.x * TT;
  const float gate = 1.0f / (1.0f + expf(-kvg[0]));
  const unsigned lds0 = (unsigned)(uintptr_t)(&es[0][0]);
  const unsigned lds1 = (unsigned)(uintptr_t)(&es[1][0]);

  // prologue: start DMA of K-expert 0 while we stage q/rw manually
  if (tid < 32) tdm_load_tile(ke, lds0);

  for (int i = tid; i < TT * C; i += 256) {
    int t = i >> 7, c = i & (C - 1);
    qs[t * PP + c] = q[(size_t)(t0 + t) * C + c];
  }
  for (int i = tid; i < TT * E; i += 256) rws[i] = rw[(size_t)t0 * E + i];

  // ---- phase A: scores ----
  const int tl = tid & 31;   // token within block (lane within wave)
  const int kq = tid >> 5;   // key slot group 0..7 (uniform per wave)
  float dyn[4] = {0.f, 0.f, 0.f, 0.f};
  float stat[4];
#pragma unroll
  for (int j = 0; j < 4; ++j) {       // static-K scores (global reads, overlap DMA)
    const int k = kq + 8 * j;
    const float* skr = sk + k * C;
    float a = 0.f;
    for (int c2 = 0; c2 < C; ++c2) a = fmaf(q[(size_t)(t0 + tl) * C + c2], skr[c2], a);
    stat[j] = a;
  }
  __syncthreads();   // qs/rws visible

  for (int e = 0; e < E; ++e) {
    if (tid < 32) {
      if (e + 1 < E) {
        tdm_load_tile(ke + (size_t)(e + 1) * KB * C, (e & 1) ? lds0 : lds1);
        __builtin_amdgcn_s_wait_tensorcnt(1);   // tile e complete (in-order)
      } else {
        __builtin_amdgcn_s_wait_tensorcnt(0);
      }
    }
    __syncthreads();                            // publish tile e to all waves
    const float* et = es[e & 1];
    const float rwe = rws[tl * E + e];
#pragma unroll
    for (int j = 0; j < 4; ++j) {
      const int k = kq + 8 * j;
      const float4* qv = (const float4*)(qs + tl * PP);
      const float4* kv = (const float4*)(et + k * PP);
      float a = 0.f;
      for (int c4 = 0; c4 < C / 4; ++c4) {
        float4 qa = qv[c4], ka = kv[c4];
        a = fmaf(qa.x, ka.x, a); a = fmaf(qa.y, ka.y, a);
        a = fmaf(qa.z, ka.z, a); a = fmaf(qa.w, ka.w, a);
      }
      dyn[j] = fmaf(rwe, a, dyn[j]);
    }
    __syncthreads();                            // all reads of tile e done
  }

  const float scale = 0.08838834764831845f;     // 1/sqrt(128)
  float my[4];
#pragma unroll
  for (int j = 0; j < 4; ++j) {
    my[j] = (gate * dyn[j] + (1.f - gate) * stat[j]) * scale;
    sc[tl * (KB + 1) + kq + 8 * j] = my[j];
  }
  __syncthreads();
  float m = -1e30f;
  for (int k = 0; k < KB; ++k) m = fmaxf(m, sc[tl * (KB + 1) + k]);
  float ssum = 0.f;
  for (int k = 0; k < KB; ++k) ssum += expf(sc[tl * (KB + 1) + k] - m);
  __syncthreads();
  const float isum = 1.0f / ssum;
#pragma unroll
  for (int j = 0; j < 4; ++j) sc[tl * (KB + 1) + kq + 8 * j] = expf(my[j] - m) * isum;

  // ---- phase B: attn-weighted V mixing ----
  if (tid < 32) tdm_load_tile(ve, lds0);        // start V-expert 0 DMA
  __syncthreads();                              // attn weights visible

  const int tv = tid >> 3;  // token
  const int cs = tid & 7;   // channel slot; owns c = cs + 8*i
  float outd[16], outs[16];
#pragma unroll
  for (int i = 0; i < 16; ++i) { outd[i] = 0.f; outs[i] = 0.f; }
  for (int k = 0; k < KB; ++k) {                // static-V part (overlaps DMA)
    const float w = sc[tv * (KB + 1) + k];
    const float* svr = sv + k * C + cs;
#pragma unroll
    for (int i = 0; i < 16; ++i) outs[i] = fmaf(w, svr[8 * i], outs[i]);
  }
  for (int e = 0; e < E; ++e) {
    if (tid < 32) {
      if (e + 1 < E) {
        tdm_load_tile(ve + (size_t)(e + 1) * KB * C, (e & 1) ? lds0 : lds1);
        __builtin_amdgcn_s_wait_tensorcnt(1);
      } else {
        __builtin_amdgcn_s_wait_tensorcnt(0);
      }
    }
    __syncthreads();
    const float* et = es[e & 1];
    float tmp[16];
#pragma unroll
    for (int i = 0; i < 16; ++i) tmp[i] = 0.f;
    for (int k = 0; k < KB; ++k) {
      const float w = sc[tv * (KB + 1) + k];
      const float* vr = et + k * PP + cs;
#pragma unroll
      for (int i = 0; i < 16; ++i) tmp[i] = fmaf(w, vr[8 * i], tmp[i]);
    }
    const float rwe = rws[tv * E + e];
#pragma unroll
    for (int i = 0; i < 16; ++i) outd[i] = fmaf(rwe, tmp[i], outd[i]);
    __syncthreads();
  }
#pragma unroll
  for (int i = 0; i < 16; ++i)
    attn_out[(size_t)(t0 + tv) * C + cs + 8 * i] = gate * outd[i] + (1.f - gate) * outs[i];
}

// ------------- f32 -> bf16 conversion -------------
__global__ void k_f32_to_bf16(const float* __restrict__ in, unsigned short* __restrict__ out, int n) {
  int i = blockIdx.x * 256 + threadIdx.x;
  if (i < n) out[i] = f2bf_rne(in[i]);
}

// ------------- LM head: [4096,128]bf16 x [50257,128]bf16^T -> f32 logits, via WMMA -------------
#define NT_TILES ((V + 15) / 16)  // 3142
#define MGROUPS  (BT / 64)        // 64 (each wave: 4 M-tiles of 16)

__global__ __launch_bounds__(256) void k_lmhead_wmma(const bf16_t* __restrict__ xb,
                                                     const bf16_t* __restrict__ wb,
                                                     float* __restrict__ out) {
  const int wave = (blockIdx.x << 3) + (threadIdx.x >> 5);
  if (wave >= MGROUPS * NT_TILES) return;   // wave-uniform: EXEC stays all-1s for WMMA
  const int nt  = wave / MGROUPS;
  const int mg  = wave - nt * MGROUPS;
  const int lane = threadIdx.x & 31;
  const int half = lane >> 4;   // 0: lanes 0-15, 1: lanes 16-31
  const int l16  = lane & 15;

  const int n  = nt * 16 + l16;
  const int nc = (n < V) ? n : (V - 1);
  // B fragment (32x16 K-major): lane half selects K {0-7,16-23} vs {8-15,24-31}
  const bf16_t* brow  = wb + (size_t)nc * C + half * 8;
  // A fragment (16x32): lane half selects K {0-7,16-23} vs {8-15,24-31}
  const bf16_t* arow0 = xb + (size_t)(mg * 64 + l16) * C + half * 8;

  v8f acc[4] = {};
#pragma unroll
  for (int kc = 0; kc < 4; ++kc) {
    v8bf blo = *(const v8bf*)(brow + kc * 32);
    v8bf bhi = *(const v8bf*)(brow + kc * 32 + 16);
    v16bf bv;
#pragma unroll
    for (int z = 0; z < 8; ++z) { bv[z] = blo[z]; bv[z + 8] = bhi[z]; }
#pragma unroll
    for (int i = 0; i < 4; ++i) {
      const bf16_t* ap = arow0 + (size_t)i * 16 * C + kc * 32;
      v8bf lo = *(const v8bf*)(ap);        // K = base+0..7
      v8bf hi = *(const v8bf*)(ap + 16);   // K = base+16..23
      v16bf av;
#pragma unroll
      for (int z = 0; z < 8; ++z) { av[z] = lo[z]; av[z + 8] = hi[z]; }
      acc[i] = __builtin_amdgcn_wmma_f32_16x16x32_bf16(
          false, av, false, bv, (short)0, acc[i], false, false);
    }
  }
  if (n < V) {
#pragma unroll
    for (int i = 0; i < 4; ++i) {
      const int rowb = mg * 64 + i * 16 + half * 8;  // C/D layout: vgpr j -> M=j (+8 for hi half)
#pragma unroll
      for (int j = 0; j < 8; ++j)
        out[(size_t)(rowb + j) * V + n] = acc[i][j];
    }
  }
}

// ---------------- host launch ----------------
extern "C" void kernel_launch(void* const* d_in, const int* in_sizes, int n_in,
                              void* d_out, int out_size, void* d_ws, size_t ws_size,
                              hipStream_t stream) {
  const int*   ids   = (const int*)  d_in[0];
  const float* emb   = (const float*)d_in[1];
  const float* router= (const float*)d_in[2];
  const float* ke    = (const float*)d_in[3];
  const float* ve    = (const float*)d_in[4];
  const float* sk    = (const float*)d_in[5];
  const float* sv    = (const float*)d_in[6];
  const float* kvg   = (const float*)d_in[7];
  const float* Wq    = (const float*)d_in[8];
  const float* Wo    = (const float*)d_in[9];
  const float* Wmlp  = (const float*)d_in[10];
  const float* bmlp  = (const float*)d_in[11];
  const float* ln1g  = (const float*)d_in[12];
  const float* ln1b  = (const float*)d_in[13];
  const float* ln2g  = (const float*)d_in[14];
  const float* ln2b  = (const float*)d_in[15];
  const float* Wlm   = (const float*)d_in[16];

  char* ws = (char*)d_ws;
  float* x    = (float*)(ws + OFS_X);
  float* xln  = (float*)(ws + OFS_XLN);
  float* rw   = (float*)(ws + OFS_RW);
  float* q    = (float*)(ws + OFS_Q);
  float* amid = (float*)(ws + OFS_ATT);
  float* xres = (float*)(ws + OFS_XRES);
  unsigned short* x2b  = (unsigned short*)(ws + OFS_X2B);
  unsigned short* wlmb = (unsigned short*)(ws + OFS_WLMB);

  float* logits   = (float*)d_out;
  float* gate_out = logits + (size_t)BT * V;

  k_embed_ln_gate<<<BT, C, 0, stream>>>(ids, emb, router, ln1g, ln1b, x, xln, rw, gate_out);
  k_rowmat<<<BT, C, 0, stream>>>(xln, Wq, nullptr, nullptr, q);
  k_attn<<<BT / TT, 256, 0, stream>>>(q, rw, ke, ve, sk, sv, kvg, amid);
  k_rowmat<<<BT, C, 0, stream>>>(amid, Wo, x, nullptr, xres);      // x + attn@Wo
  k_layernorm<<<BT, C, 0, stream>>>(xres, ln2g, ln2b, xln);        // reuse xln buffer
  k_rowmat<<<BT, C, 0, stream>>>(xln, Wmlp, xres, bmlp, x);        // x2 into x buffer
  k_f32_to_bf16<<<(BT * C + 255) / 256, 256, 0, stream>>>(x, x2b, BT * C);
  k_f32_to_bf16<<<(V * C + 255) / 256, 256, 0, stream>>>(Wlm, wlmb, V * C);

  const int waves  = MGROUPS * NT_TILES;
  const int blocks = (waves + 7) / 8;
  k_lmhead_wmma<<<blocks, 256, 0, stream>>>((const bf16_t*)x2b, (const bf16_t*)wlmb, logits);
}
